// MultiTaskGCN_37623913513359
// MI455X (gfx1250) — matrix operand order repaired and checked by reading, not compiled, not run on hardware
//
#include <hip/hip_runtime.h>

typedef __bf16 bf16_t;
typedef __attribute__((ext_vector_type(16))) __bf16 v16bf;
typedef __attribute__((ext_vector_type(8)))  __bf16 v8bf;
typedef __attribute__((ext_vector_type(8)))  float  v8f;
typedef __attribute__((ext_vector_type(4)))  unsigned int v4u;
typedef __attribute__((ext_vector_type(8)))  int v8i;
typedef __attribute__((ext_vector_type(4)))  int v4i;

#define N_NODES 100000
#define F_IN    512
#define H_DIM   256
#define C_PAD   64     // padded 40+16+1
#define C_REAL  57

static __device__ __forceinline__ bf16_t f2bf(float f) {
    unsigned u = __builtin_bit_cast(unsigned, f);
    u += 0x7FFFu + ((u >> 16) & 1u);          // round-to-nearest-even
    unsigned short hs = (unsigned short)(u >> 16);
    return __builtin_bit_cast(bf16_t, hs);
}

// ---------------- degree / norm ----------------
__global__ void zero_f32_kernel(float* p, int n) {
    int i = blockIdx.x * blockDim.x + threadIdx.x;
    if (i < n) p[i] = 0.0f;
}

__global__ void deg_count_kernel(const int* __restrict__ ei, int E, float* deg) {
    int e = blockIdx.x * blockDim.x + threadIdx.x;
    if (e < E) {
        int d = ei[E + e];
        unsafeAtomicAdd(&deg[d], 1.0f);
    }
}

__global__ void deg_finalize_kernel(float* dinv, int n) {
    int i = blockIdx.x * blockDim.x + threadIdx.x;
    if (i < n) dinv[i] = rsqrtf(dinv[i] + 1.0f);   // +1 self loop; always > 0
}

// ---------------- weight packing (transpose to [Ncol][K] bf16) ----------------
__global__ void pack_w1_kernel(const float* __restrict__ W1, bf16_t* __restrict__ W1t) {
    int idx = blockIdx.x * blockDim.x + threadIdx.x;   // over 512*256
    if (idx < F_IN * H_DIM) {
        int k = idx / H_DIM, n = idx % H_DIM;          // W1 is [K=512][N=256]
        W1t[n * F_IN + k] = f2bf(W1[idx]);
    }
}

__global__ void pack_wcat_kernel(const float* __restrict__ Wc, const float* __restrict__ Wk,
                                 const float* __restrict__ Wd, bf16_t* __restrict__ Wt) {
    int idx = blockIdx.x * blockDim.x + threadIdx.x;   // over 64*256
    if (idx >= C_PAD * H_DIM) return;
    int n = idx >> 8, k = idx & 255;
    float v = 0.0f;
    if (n < 40)       v = Wc[k * 40 + n];
    else if (n < 56)  v = Wk[k * 16 + (n - 40)];
    else if (n == 56) v = Wd[k];
    Wt[n * H_DIM + k] = f2bf(v);
}

// ---------------- WMMA GEMM: C[M,Ncols] = (relu?)A[M,K]fp32 @ Bt[Ncols,K]bf16 ----------------
// Block = 4 waves, each wave computes a 16 x (NT*16) strip.
// B panel ([NT*16][K] bf16, contiguous in Bt) is staged into LDS once per block,
// via the Tensor Data Mover (6-arg builtin on this toolchain).
template <int NT>
__global__ __launch_bounds__(128)
void wmma_gemm_kernel(const float* __restrict__ A, int lda,
                      const bf16_t* __restrict__ Bt,
                      float* __restrict__ C, int ldc,
                      int M, int K, int relu_a)
{
    extern __shared__ __align__(16) bf16_t lds_b[];   // [NT*16][K]
    const int lane = threadIdx.x & 31;
    const int wave = threadIdx.x >> 5;
    const int half = lane >> 4;
    const int r    = lane & 15;
    const int m0   = (blockIdx.x * 4 + wave) * 16;
    const int nbase = blockIdx.y * (NT * 16);
    const int rows  = NT * 16;

    // ---- stage B panel into LDS via TDM ----
#if __has_builtin(__builtin_amdgcn_tensor_load_to_lds) && __has_builtin(__builtin_amdgcn_s_wait_tensorcnt)
    if (wave == 0) {
        unsigned long long gaddr = (unsigned long long)(const void*)(Bt + (size_t)nbase * K);
        unsigned lds_off = (unsigned)(unsigned long long)(void*)lds_b;
        // D# group 0: count=1 | lds_addr | global_addr[56:0] | type=2
        v4u g0;
        g0[0] = 1u;
        g0[1] = lds_off;
        g0[2] = (unsigned)(gaddr & 0xffffffffu);
        g0[3] = (unsigned)((gaddr >> 32) & 0x01ffffffu) | (2u << 30);
        // D# group 1: wg_mask=0, data_size=1 (2B), 2D tensor == tile: [K x rows], stride K
        v8i g1;
        g1[0] = (int)(1u << 16);                                   // data_size = 2 bytes
        g1[1] = (int)(((unsigned)K & 0xffffu) << 16);              // tensor_dim0[15:0]
        g1[2] = (int)((((unsigned)K >> 16) & 0xffffu) |
                      (((unsigned)rows & 0xffffu) << 16));         // dim0[31:16] | tensor_dim1[15:0]
        g1[3] = (int)(((unsigned)K & 0xffffu) << 16);              // dim1[31:16]=0 | tile_dim0
        g1[4] = (int)((unsigned)rows & 0xffffu);                   // tile_dim1 | tile_dim2=0
        g1[5] = (int)K;                                            // tensor_dim0_stride[31:0]
        g1[6] = 0;                                                 // stride[47:32] | dim1_stride lo
        g1[7] = 0;
        v4i g2 = {0, 0, 0, 0};                                     // 2D: groups 2/3 disabled
        v4i g3 = {0, 0, 0, 0};
        v8i g4 = {0, 0, 0, 0, 0, 0, 0, 0};
        __builtin_amdgcn_tensor_load_to_lds(g0, g1, g2, g3, g4, 0);
        __builtin_amdgcn_s_wait_tensorcnt(0);
    }
#else
    {
        const v8bf* gsrc = (const v8bf*)(Bt + (size_t)nbase * K);
        v8bf* ldst = (v8bf*)lds_b;
        const int chunks = rows * K / 8;
        for (int i = threadIdx.x; i < chunks; i += blockDim.x) ldst[i] = gsrc[i];
    }
#endif
    __syncthreads();

    v8f acc[NT];
#pragma unroll
    for (int t = 0; t < NT; ++t) acc[t] = (v8f){0.f,0.f,0.f,0.f,0.f,0.f,0.f,0.f};

    if (m0 < M) {
        const float* arow = A + (size_t)(m0 + r) * lda;
        for (int kk = 0; kk < K; kk += 32) {
            // A fragment (16x32 bf16): lane half h holds K = j + 8*(h + j/8), j=0..15
            const float* ap = arow + kk + 8 * half;
            float4 a0 = *(const float4*)(ap + 0);
            float4 a1 = *(const float4*)(ap + 4);
            float4 a2 = *(const float4*)(ap + 16);
            float4 a3 = *(const float4*)(ap + 20);
            float av[16] = {a0.x,a0.y,a0.z,a0.w, a1.x,a1.y,a1.z,a1.w,
                            a2.x,a2.y,a2.z,a2.w, a3.x,a3.y,a3.z,a3.w};
            v16bf afrag;
#pragma unroll
            for (int j = 0; j < 16; ++j) {
                float f = av[j];
                if (relu_a) f = fmaxf(f, 0.0f);
                afrag[j] = f2bf(f);
            }
            // fetch all B fragments first (ds_load_b128 clause), then WMMAs back-to-back
            v16bf bfr[NT];
#pragma unroll
            for (int t = 0; t < NT; ++t) {
                const bf16_t* bp = lds_b + (size_t)(t * 16 + r) * K + kk + 8 * half;
                v8bf blo = *(const v8bf*)bp;
                v8bf bhi = *(const v8bf*)(bp + 16);
#pragma unroll
                for (int j = 0; j < 8; ++j) { bfr[t][j] = blo[j]; bfr[t][j + 8] = bhi[j]; }
            }
#pragma unroll
            for (int t = 0; t < NT; ++t) {
                acc[t] = __builtin_amdgcn_wmma_f32_16x16x32_bf16(
                    false, afrag, false, bfr[t], (short)0, acc[t], false, false);
            }
        }

        // C/D layout: VGPR v -> row m0 + v + 8*half, col = nbase + t*16 + r
#pragma unroll
        for (int t = 0; t < NT; ++t) {
            int col = nbase + t * 16 + r;
#pragma unroll
            for (int v = 0; v < 8; ++v) {
                int row = m0 + v + 8 * half;
                C[(size_t)row * ldc + col] = acc[t][v];
            }
        }
    }
}

// ---------------- layer-1 propagation ----------------
__global__ void init_h_kernel(const float* __restrict__ hw, const float* __restrict__ dinv,
                              const float* __restrict__ b1, float* __restrict__ h)
{
    int idx = blockIdx.x * blockDim.x + threadIdx.x;      // N*64
    if (idx >= N_NODES * 64) return;
    int n = idx >> 6, c4 = (idx & 63) << 2;
    float s = dinv[n]; s = s * s;
    float4 v = *(const float4*)(hw + (size_t)n * H_DIM + c4);
    float4 b = *(const float4*)(b1 + c4);
    float4 o = make_float4(b.x + s * v.x, b.y + s * v.y, b.z + s * v.z, b.w + s * v.w);
    *(float4*)(h + (size_t)n * H_DIM + c4) = o;
}

__global__ __launch_bounds__(256)
void edge_prop_h_kernel(const int* __restrict__ ei, int E,
                        const float* __restrict__ hw, const float* __restrict__ dinv,
                        float* __restrict__ h)
{
    int lane = threadIdx.x & 31;
    int wid = blockIdx.x * (blockDim.x >> 5) + (threadIdx.x >> 5);
    int stride = gridDim.x * (blockDim.x >> 5);
    for (int e = wid; e < E; e += stride) {
        int s = ei[e], d = ei[E + e];
        int en = e + stride;
        if (en < E) {
            int sn = ei[en];
            __builtin_prefetch(hw + (size_t)sn * H_DIM + lane * 8, 0, 3);
        }
        float w = dinv[s] * dinv[d];
        const float* sp = hw + (size_t)s * H_DIM + lane * 8;
        float*       dp = h  + (size_t)d * H_DIM + lane * 8;
        float4 v0 = *(const float4*)(sp + 0);
        float4 v1 = *(const float4*)(sp + 4);
        unsafeAtomicAdd(dp + 0, w * v0.x);
        unsafeAtomicAdd(dp + 1, w * v0.y);
        unsafeAtomicAdd(dp + 2, w * v0.z);
        unsafeAtomicAdd(dp + 3, w * v0.w);
        unsafeAtomicAdd(dp + 4, w * v1.x);
        unsafeAtomicAdd(dp + 5, w * v1.y);
        unsafeAtomicAdd(dp + 6, w * v1.z);
        unsafeAtomicAdd(dp + 7, w * v1.w);
    }
}

// ---------------- head propagation (fused 40+16+1 = 57 cols, directly into d_out) ----------------
static __device__ __forceinline__ int out_index(int n, int c) {
    if (c < 40) return n * 40 + c;
    if (c < 56) return 40 * N_NODES + n * 16 + (c - 40);
    return 56 * N_NODES + n;
}

__global__ void init_out_kernel(const float* __restrict__ hcat, const float* __restrict__ dinv,
                                const float* __restrict__ bc, const float* __restrict__ bk,
                                const float* __restrict__ bd, float* __restrict__ out)
{
    int idx = blockIdx.x * blockDim.x + threadIdx.x;      // N*57
    if (idx >= N_NODES * C_REAL) return;
    int n = idx / C_REAL, c = idx % C_REAL;
    float bias = (c < 40) ? bc[c] : (c < 56) ? bk[c - 40] : bd[0];
    float s = dinv[n]; s = s * s;
    out[out_index(n, c)] = bias + s * hcat[(size_t)n * C_PAD + c];
}

__global__ __launch_bounds__(256)
void edge_prop_out_kernel(const int* __restrict__ ei, int E,
                          const float* __restrict__ hcat, const float* __restrict__ dinv,
                          float* __restrict__ out)
{
    int lane = threadIdx.x & 31;
    int wid = blockIdx.x * (blockDim.x >> 5) + (threadIdx.x >> 5);
    int stride = gridDim.x * (blockDim.x >> 5);
    for (int e = wid; e < E; e += stride) {
        int s = ei[e], d = ei[E + e];
        int en = e + stride;
        if (en < E) {
            int sn = ei[en];
            __builtin_prefetch(hcat + (size_t)sn * C_PAD + lane, 0, 3);
        }
        float w = dinv[s] * dinv[d];
        const float* sp = hcat + (size_t)s * C_PAD;
        float v0 = sp[lane];
        unsafeAtomicAdd(&out[out_index(d, lane)], w * v0);
        if (lane < C_REAL - 32) {
            float v1 = sp[lane + 32];
            unsafeAtomicAdd(&out[out_index(d, lane + 32)], w * v1);
        }
    }
}

// ---------------- launch ----------------
static inline size_t align_up(size_t x) { return (x + 255) & ~(size_t)255; }

extern "C" void kernel_launch(void* const* d_in, const int* in_sizes, int n_in,
                              void* d_out, int out_size, void* d_ws, size_t ws_size,
                              hipStream_t stream)
{
    const float* x  = (const float*)d_in[0];
    const int*   ei = (const int*)  d_in[1];
    const float* W1 = (const float*)d_in[2];
    const float* b1 = (const float*)d_in[3];
    const float* Wc = (const float*)d_in[4];
    const float* bc = (const float*)d_in[5];
    const float* Wk = (const float*)d_in[6];
    const float* bk = (const float*)d_in[7];
    const float* Wd = (const float*)d_in[8];
    const float* bd = (const float*)d_in[9];
    float* out = (float*)d_out;
    const int E = in_sizes[1] / 2;
    const int N = N_NODES;

    char* ws = (char*)d_ws;
    float*  dinv = (float*)ws;  ws += align_up((size_t)N * 4);
    bf16_t* W1t  = (bf16_t*)ws; ws += align_up((size_t)F_IN * H_DIM * 2);
    bf16_t* Wct  = (bf16_t*)ws; ws += align_up((size_t)C_PAD * H_DIM * 2);
    float*  hw   = (float*)ws;  ws += align_up((size_t)N * H_DIM * 4);
    float*  h    = (float*)ws;  ws += align_up((size_t)N * H_DIM * 4);
    float*  hcat = (float*)ws;  ws += align_up((size_t)N * C_PAD * 4);

    // 1. symmetric norm: dinv = rsqrt(indeg + 1)
    zero_f32_kernel<<<(N + 255) / 256, 256, 0, stream>>>(dinv, N);
    deg_count_kernel<<<(E + 255) / 256, 256, 0, stream>>>(ei, E, dinv);
    deg_finalize_kernel<<<(N + 255) / 256, 256, 0, stream>>>(dinv, N);

    // 2. pack weights -> transposed bf16
    pack_w1_kernel<<<(F_IN * H_DIM + 255) / 256, 256, 0, stream>>>(W1, W1t);
    pack_wcat_kernel<<<(C_PAD * H_DIM + 255) / 256, 256, 0, stream>>>(Wc, Wk, Wd, Wct);

    // 3. hw = x @ W1  (WMMA bf16; B panel 128x512 bf16 = 128KB LDS via TDM)
    const int mtiles = N / 16;                         // 6250
    dim3 g1((mtiles + 3) / 4, H_DIM / 128);            // 2 col groups of 128
    wmma_gemm_kernel<8><<<g1, 128, (size_t)(8 * 16) * F_IN * 2, stream>>>(
        x, F_IN, W1t, hw, H_DIM, N, F_IN, 0);

    // 4. h = bias + self + scatter-add(norm * hw[src])   (ReLU deferred to next GEMM read)
    init_h_kernel<<<(N * 64 + 255) / 256, 256, 0, stream>>>(hw, dinv, b1, h);
    edge_prop_h_kernel<<<16384, 256, 0, stream>>>(ei, E, hw, dinv, h);

    // 5. hcat = relu(h) @ [Wc|Wk|Wd]  (fused heads; B panel 64x256 bf16 = 32KB LDS)
    dim3 g2((mtiles + 3) / 4, 1);
    wmma_gemm_kernel<4><<<g2, 128, (size_t)(4 * 16) * H_DIM * 2, stream>>>(
        h, H_DIM, Wct, hcat, C_PAD, N, H_DIM, 1);

    // 6. heads: bias + self + scatter-add directly into d_out (57-wide)
    init_out_kernel<<<(N * C_REAL + 255) / 256, 256, 0, stream>>>(hcat, dinv, bc, bk, bd, out);
    edge_prop_out_kernel<<<16384, 256, 0, stream>>>(ei, E, hcat, dinv, out);
}
